// GNNModel_76665166233741
// MI455X (gfx1250) — matrix operand order repaired and checked by reading, not compiled, not run on hardware
//
#include <hip/hip_runtime.h>
#include <hip/hip_bf16.h>

typedef __bf16 bf16;
typedef __attribute__((ext_vector_type(16))) __bf16 v16bf;
typedef __attribute__((ext_vector_type(8)))  __bf16 v8bf;
typedef __attribute__((ext_vector_type(8)))  float  v8f;

#define NEG_SLOPE 0.2f

// ---------------- elementwise / conversion kernels ----------------

__global__ void k_f32_to_bf16(const float* __restrict__ in, bf16* __restrict__ out, int n) {
  int i = blockIdx.x * blockDim.x + threadIdx.x;
  if (i < n) out[i] = (bf16)in[i];
}

// Wt[n*K + k] = (bf16) W[k*N + n]   (pre-transpose so B tile loads are contiguous)
__global__ void k_transpose_bf16(const float* __restrict__ W, bf16* __restrict__ Wt, int K, int N) {
  int i = blockIdx.x * blockDim.x + threadIdx.x;
  if (i >= K * N) return;
  int k = i / N, n = i % N;
  Wt[(size_t)n * K + k] = (bf16)W[i];
}

__global__ void k_fill(float* __restrict__ p, float v, int n) {
  int i = blockIdx.x * blockDim.x + threadIdx.x;
  if (i < n) p[i] = v;
}

// ---------------- WMMA GEMM: C[M,N] = A[M,K] * B[K,N] ----------------
// A bf16 row-major [M,K]; Bt = B^T bf16 [N,K] row-major; C f32 [M,N].
// One wave computes a 16x64 strip: 1 A fragment reused across 4 B fragments
// per k-step (4 back-to-back v_wmma sharing A -> 12.8 FLOP/B from L2).

__global__ void __launch_bounds__(256)
k_wmma_gemm(const bf16* __restrict__ A, const bf16* __restrict__ Bt,
            float* __restrict__ C, int M, int N, int K) {
  int lane   = threadIdx.x & 31;
  int wave   = (int)((blockIdx.x * blockDim.x + threadIdx.x) >> 5);
  int tilesN4 = N >> 6;          // groups of four 16-wide N tiles
  int tilesM  = M >> 4;
  int tm = wave / tilesN4;       // M tile
  int tg = wave - tm * tilesN4;  // N group (64 columns)
  if (tm >= tilesM) return;

  int half = lane >> 4;          // 0 or 1
  int l15  = lane & 15;

  const bf16* arow = A + (size_t)(tm * 16 + l15) * K;
  const bf16* brow0 = Bt + (size_t)(tg * 64 +  0 + l15) * K;
  const bf16* brow1 = Bt + (size_t)(tg * 64 + 16 + l15) * K;
  const bf16* brow2 = Bt + (size_t)(tg * 64 + 32 + l15) * K;
  const bf16* brow3 = Bt + (size_t)(tg * 64 + 48 + l15) * K;

  v8f acc0 = {}, acc1 = {}, acc2 = {}, acc3 = {};

  for (int k0 = 0; k0 < K; k0 += 32) {
    // A lane data: K = k0 + 16*half + [0..15] (contiguous 32 bytes)
    v16bf a = *(const v16bf*)(arow + k0 + 16 * half);
    // B lane data per tile: K = k0 + 16*half + [0..15] (contiguous 32 bytes)
    int boff = k0 + 16 * half;
    v16bf b0 = *(const v16bf*)(brow0 + boff);
    v16bf b1 = *(const v16bf*)(brow1 + boff);
    v16bf b2 = *(const v16bf*)(brow2 + boff);
    v16bf b3 = *(const v16bf*)(brow3 + boff);
    acc0 = __builtin_amdgcn_wmma_f32_16x16x32_bf16(false, a, false, b0, (short)0, acc0, false, false);
    acc1 = __builtin_amdgcn_wmma_f32_16x16x32_bf16(false, a, false, b1, (short)0, acc1, false, false);
    acc2 = __builtin_amdgcn_wmma_f32_16x16x32_bf16(false, a, false, b2, (short)0, acc2, false, false);
    acc3 = __builtin_amdgcn_wmma_f32_16x16x32_bf16(false, a, false, b3, (short)0, acc3, false, false);
  }

  // C/D layout: lane -> col (lane&15), VGPR v -> row v + 8*half
  float* crow = C + (size_t)(tm * 16 + 8 * half) * N + (tg * 64 + l15);
#pragma unroll
  for (int v = 0; v < 8; ++v) {
    float* r = crow + (size_t)v * N;
    r[0]  = acc0[v];
    r[16] = acc1[v];
    r[32] = acc2[v];
    r[48] = acc3[v];
  }
}

// ---------------- per-node attention logits: es = H . a_s, ed = H . a_d ----------------

__global__ void k_node_attn(const float* __restrict__ H, const float* __restrict__ as_,
                            const float* __restrict__ ad_, float* __restrict__ es,
                            float* __restrict__ ed, int N, int F) {
  int wave = (int)((blockIdx.x * blockDim.x + threadIdx.x) >> 5);
  int lane = threadIdx.x & 31;
  if (wave >= N) return;
  const float* h = H + (size_t)wave * F;
  float ss = 0.f, sd = 0.f;
  for (int f = lane; f < F; f += 32) {
    float hv = h[f];
    ss += hv * as_[f];
    sd += hv * ad_[f];
  }
  for (int off = 16; off > 0; off >>= 1) {
    ss += __shfl_down(ss, off);
    sd += __shfl_down(sd, off);
  }
  if (lane == 0) { es[wave] = ss; ed[wave] = sd; }
}

// ---------------- edge kernels ----------------

__device__ __forceinline__ void atomicMaxF(float* addr, float val) {
  int* ai = (int*)addr;
  int old = __float_as_int(*addr);
  while (__int_as_float(old) < val) {
    int assumed = old;
    old = atomicCAS(ai, assumed, __float_as_int(val));
    if (old == assumed) break;
  }
}

// e_val = leaky_relu(es[src] + ed[dst]); segment_max into maxb[dst]
__global__ void k_edge_logits(const int* __restrict__ ei, int E, int Etot,
                              const float* __restrict__ es, const float* __restrict__ ed,
                              float* __restrict__ ev, float* __restrict__ maxb) {
  int e = blockIdx.x * blockDim.x + threadIdx.x;
  if (e >= Etot) return;
  int src = (e < E) ? ei[e]     : (e - E);   // self loops appended
  int dst = (e < E) ? ei[E + e] : (e - E);
  float v = es[src] + ed[dst];
  v = (v > 0.f) ? v : NEG_SLOPE * v;
  ev[e] = v;
  atomicMaxF(&maxb[dst], v);
}

// ex = exp(e - max[dst]); segment_sum into den[dst]
__global__ void k_edge_exp(const int* __restrict__ ei, int E, int Etot,
                           const float* __restrict__ ev, const float* __restrict__ maxb,
                           float* __restrict__ ex, float* __restrict__ den) {
  int e = blockIdx.x * blockDim.x + threadIdx.x;
  if (e >= Etot) return;
  int dst = (e < E) ? ei[E + e] : (e - E);
  float v = __expf(ev[e] - maxb[dst]);
  ex[e] = v;
  atomicAdd(&den[dst], v);
}

// out[dst] += alpha * H[src]; one wave per edge, lanes over features
__global__ void __launch_bounds__(256)
k_edge_scatter(const int* __restrict__ ei, int E, int Etot,
               const float* __restrict__ H, const float* __restrict__ ex,
               const float* __restrict__ den, float* __restrict__ out, int F) {
  int wave = (int)((blockIdx.x * blockDim.x + threadIdx.x) >> 5);
  int lane = threadIdx.x & 31;
  if (wave >= Etot) return;
  int src = (wave < E) ? ei[wave]     : (wave - E);
  int dst = (wave < E) ? ei[E + wave] : (wave - E);
  float alpha = ex[wave] / den[dst];
  const float* h = H + (size_t)src * F;
  float* o = out + (size_t)dst * F;
  for (int f = lane; f < F; f += 32) atomicAdd(&o[f], alpha * h[f]);
}

// out = relu(out + b); optionally emit bf16 copy for the next layer's GEMM A
__global__ void k_bias_relu(float* __restrict__ H, const float* __restrict__ b,
                            bf16* __restrict__ Abf, int N, int F) {
  size_t i = (size_t)blockIdx.x * blockDim.x + threadIdx.x;
  if (i >= (size_t)N * F) return;
  int f = (int)(i % F);
  float v = H[i] + b[f];
  v = (v > 0.f) ? v : 0.f;
  H[i] = v;
  if (Abf) Abf[i] = (bf16)v;
}

// ---------------- pooling + MLP head ----------------

__global__ void k_pool_max(const float* __restrict__ H, float* __restrict__ pooled,
                           int N, int F) {
  int f = blockIdx.x * blockDim.x + threadIdx.x;
  if (f >= F) return;
  float m = H[f];
  for (int i = 1; i < N; ++i) {
    float v = H[(size_t)i * F + f];
    m = (v > m) ? v : m;
  }
  pooled[f] = m;
}

__global__ void __launch_bounds__(512)
k_head(const float* __restrict__ pooled,
       const float* __restrict__ lw1, const float* __restrict__ lb1,
       const float* __restrict__ lw2, const float* __restrict__ lb2,
       float* __restrict__ out) {
  __shared__ float z1[512];
  __shared__ float z2[10];
  int j = threadIdx.x;
  float s = lb1[j];
  for (int k = 0; k < 1024; ++k) s += pooled[k] * lw1[k * 512 + j];
  z1[j] = (s > 0.f) ? s : 0.f;
  __syncthreads();
  if (j < 10) {
    float t = lb2[j];
    for (int k = 0; k < 512; ++k) t += z1[k] * lw2[k * 10 + j];
    z2[j] = t;
  }
  __syncthreads();
  if (j == 0) {
    float m = z2[0];
    for (int i = 1; i < 10; ++i) m = (z2[i] > m) ? z2[i] : m;
    float acc = 0.f;
    for (int i = 0; i < 10; ++i) acc += __expf(z2[i] - m);
    float lse = m + __logf(acc);
    for (int i = 0; i < 10; ++i) out[i] = z2[i] - lse;
  }
}

// ---------------- host launcher ----------------

static inline char* carve(char*& p, size_t bytes) {
  char* r = p;
  p += (bytes + 255) & ~(size_t)255;
  return r;
}

extern "C" void kernel_launch(void* const* d_in, const int* in_sizes, int n_in,
                              void* d_out, int out_size, void* d_ws, size_t ws_size,
                              hipStream_t stream) {
  const float* x  = (const float*)d_in[0];
  const int*   ei = (const int*)d_in[1];
  // d_in[2] = edge_attr: ignored by reference
  const float* W[3]  = {(const float*)d_in[3],  (const float*)d_in[7],  (const float*)d_in[11]};
  const float* AS[3] = {(const float*)d_in[4],  (const float*)d_in[8],  (const float*)d_in[12]};
  const float* AD[3] = {(const float*)d_in[5],  (const float*)d_in[9],  (const float*)d_in[13]};
  const float* B[3]  = {(const float*)d_in[6],  (const float*)d_in[10], (const float*)d_in[14]};
  const float* lw1 = (const float*)d_in[15];
  const float* lb1 = (const float*)d_in[16];
  const float* lw2 = (const float*)d_in[17];
  const float* lb2 = (const float*)d_in[18];

  const int N    = in_sizes[0] / 128;   // 10000
  const int E    = in_sizes[1] / 2;     // 160000
  const int Etot = E + N;               // + self loops

  const int Fin[3]  = {128, 64, 256};
  const int Fout[3] = {64, 256, 1024};

  // workspace carving
  char* p = (char*)d_ws;
  bf16*  ABF    = (bf16*) carve(p, (size_t)N * 256 * sizeof(bf16));     // activations bf16 (max width 256)
  bf16*  WT     = (bf16*) carve(p, (size_t)256 * 1024 * sizeof(bf16));  // W^T bf16 (max 256x1024)
  float* H      = (float*)carve(p, (size_t)N * 1024 * sizeof(float));   // h = x@W
  float* OUT    = (float*)carve(p, (size_t)N * 1024 * sizeof(float));   // aggregated output
  float* es     = (float*)carve(p, (size_t)N * sizeof(float));
  float* ed     = (float*)carve(p, (size_t)N * sizeof(float));
  float* maxb   = (float*)carve(p, (size_t)N * sizeof(float));
  float* den    = (float*)carve(p, (size_t)N * sizeof(float));
  float* ev     = (float*)carve(p, (size_t)Etot * sizeof(float));
  float* ex     = (float*)carve(p, (size_t)Etot * sizeof(float));
  float* pooled = (float*)carve(p, 1024 * sizeof(float));

  // layer-1 input -> bf16
  {
    int n = N * 128;
    k_f32_to_bf16<<<(n + 255) / 256, 256, 0, stream>>>(x, ABF, n);
  }

  for (int L = 0; L < 3; ++L) {
    const int K = Fin[L], F = Fout[L];

    // W^T bf16
    int kn = K * F;
    k_transpose_bf16<<<(kn + 255) / 256, 256, 0, stream>>>(W[L], WT, K, F);

    // H = A @ W  (WMMA), one wave per 16x64 strip (4 N-tiles, shared A fragment)
    int waves = (N / 16) * (F / 64);
    k_wmma_gemm<<<(waves + 7) / 8, 256, 0, stream>>>(ABF, WT, H, N, F, K);

    // attention logits per node
    k_node_attn<<<(N + 7) / 8, 256, 0, stream>>>(H, AS[L], AD[L], es, ed, N, F);

    // segment softmax over incoming edges
    k_fill<<<(N + 255) / 256, 256, 0, stream>>>(maxb, -3.0e38f, N);
    hipMemsetAsync(den, 0, (size_t)N * sizeof(float), stream);
    k_edge_logits<<<(Etot + 255) / 256, 256, 0, stream>>>(ei, E, Etot, es, ed, ev, maxb);
    k_edge_exp<<<(Etot + 255) / 256, 256, 0, stream>>>(ei, E, Etot, ev, maxb, ex, den);

    // aggregate: OUT[dst] += alpha * H[src]
    hipMemsetAsync(OUT, 0, (size_t)N * F * sizeof(float), stream);
    k_edge_scatter<<<(Etot + 7) / 8, 256, 0, stream>>>(ei, E, Etot, H, ex, den, OUT, F);

    // bias + relu (+ bf16 copy for next layer's GEMM input)
    size_t nf = (size_t)N * F;
    k_bias_relu<<<(unsigned)((nf + 255) / 256), 256, 0, stream>>>(
        OUT, B[L], (L < 2) ? ABF : (bf16*)nullptr, N, F);
  }

  // global max pool over nodes -> [1024]
  k_pool_max<<<4, 256, 0, stream>>>(OUT, pooled, N, 1024);

  // dense head + log_softmax -> d_out [10]
  k_head<<<1, 512, 0, stream>>>(pooled, lw1, lb1, lw2, lb2, (float*)d_out);
}
